// GraphFusionEncoder_27255862460867
// MI455X (gfx1250) — matrix operand order repaired and checked by reading, not compiled, not run on hardware
//
#include <hip/hip_runtime.h>

typedef __bf16 bf16;
typedef __attribute__((ext_vector_type(16))) __bf16 bf16x16;
typedef __attribute__((ext_vector_type(8)))  __bf16 bf16x8;
typedef __attribute__((ext_vector_type(4)))  __bf16 bf16x4;
typedef __attribute__((ext_vector_type(8)))  float  v8f;

static __device__ __forceinline__ v8f vzero() {
  v8f z = {0.f,0.f,0.f,0.f,0.f,0.f,0.f,0.f};
  return z;
}

// fp32 -> bf16, round-to-nearest-even
static __device__ __forceinline__ bf16 f2bf(float f) {
  unsigned u = __float_as_uint(f);
  unsigned r = u + 0x7FFFu + ((u >> 16) & 1u);
  unsigned short h = (unsigned short)(r >> 16);
  return __builtin_bit_cast(bf16, h);
}

// A-matrix 16x32 bf16 fragment from a row-major source (ld elems per row).
// Lane<16: row=lane,    K = {0..7, 16..23}
// Lane>=16: row=lane-16, K = {8..15, 24..31}
static __device__ __forceinline__ bf16x16 load_a_frag(const bf16* base, int ld, int lane) {
  int row = lane & 15;
  int k0  = (lane & 16) ? 8 : 0;
  const bf16* p = base + (size_t)row * ld + k0;
  bf16x16 a;
  ((bf16x8*)&a)[0] = *(const bf16x8*)(p);
  ((bf16x8*)&a)[1] = *(const bf16x8*)(p + 16);
  return a;
}

// B-matrix 32x16 bf16 fragment from a "transposed" source S[n][k] (row stride ld):
// B[k][n] = S[n][k]. Lane<16: col=lane, K={0..7, 16..23} ; Lane>=16: col=lane-16, K={8..15, 24..31}
static __device__ __forceinline__ bf16x16 load_bt_frag(const bf16* base, int ld, int lane) {
  int col = lane & 15;
  int k0  = (lane & 16) ? 8 : 0;
  const bf16* p = base + (size_t)col * ld + k0;
  bf16x16 b;
  ((bf16x8*)&b)[0] = *(const bf16x8*)(p);
  ((bf16x8*)&b)[1] = *(const bf16x8*)(p + 16);
  return b;
}

static __device__ __forceinline__ v8f wmma_bf16(bf16x16 a, bf16x16 b, v8f c) {
  return __builtin_amdgcn_wmma_f32_16x16x32_bf16(false, a, false, b, (short)0, c, false, false);
}

// async global->LDS, 16B per lane, tracked by ASYNCcnt (CDNA5 ISA)
static __device__ __forceinline__ void async_ld_b128(unsigned lds_byte_off, const bf16* gptr) {
  asm volatile("global_load_async_to_lds_b128 %0, %1, off"
               :: "v"(lds_byte_off), "v"((unsigned long long)(uintptr_t)gptr)
               : "memory");
}
static __device__ __forceinline__ void wait_async16() {
  asm volatile("s_wait_asynccnt 0x10" ::: "memory");
}
static __device__ __forceinline__ void wait_async0() {
  asm volatile("s_wait_asynccnt 0x0" ::: "memory");
}

// ---------------------------------------------------------------- cast kernel
__global__ void cast_bf16_kernel(const float* __restrict__ in, bf16* __restrict__ out, int n4) {
  int i = blockIdx.x * blockDim.x + threadIdx.x;
  if (i >= n4) return;
  float4 v = ((const float4*)in)[i];
  bf16x4 o;
  o[0] = f2bf(v.x); o[1] = f2bf(v.y); o[2] = f2bf(v.z); o[3] = f2bf(v.w);
  ((bf16x4*)out)[i] = o;
}

// ---------------------------------------------------------------- Q = X * W^T
// X: [M,256] row-major bf16, W: [256,256] row-major [e][d], Q: [M,256] bf16.
// Each wave owns TWO 16-row M-tiles so every W B-fragment feeds 2 WMMAs.
__global__ void __launch_bounds__(128) qproj_kernel(const bf16* __restrict__ X,
                                                    const bf16* __restrict__ W,
                                                    bf16* __restrict__ Q) {
  const int lane = threadIdx.x & 31;
  const int wave = threadIdx.x >> 5;
  const size_t row0 = ((size_t)blockIdx.x * 4 + wave) * 32;   // 32 rows per wave

  bf16x16 afr[2][8];
#pragma unroll
  for (int t = 0; t < 2; ++t)
#pragma unroll
    for (int kc = 0; kc < 8; ++kc)
      afr[t][kc] = load_a_frag(X + (row0 + t * 16) * 256 + kc * 32, 256, lane);

  const int moff = (lane & 16) ? 8 : 0;
  const int c16  = lane & 15;
#pragma unroll
  for (int nt = 0; nt < 16; ++nt) {
    v8f a0 = vzero(), a1 = vzero();
#pragma unroll
    for (int kc = 0; kc < 8; ++kc) {
      bf16x16 bf = load_bt_frag(W + (size_t)(nt * 16) * 256 + kc * 32, 256, lane);
      a0 = wmma_bf16(afr[0][kc], bf, a0);
      a1 = wmma_bf16(afr[1][kc], bf, a1);
    }
#pragma unroll
    for (int r = 0; r < 8; ++r) {
      Q[(row0 + moff + r) * 256 + nt * 16 + c16]      = f2bf(a0[r]);
      Q[(row0 + 16 + moff + r) * 256 + nt * 16 + c16] = f2bf(a1[r]);
    }
  }
}

// ------------------------------------------- fused flash attention + out-proj
// Q:[B,1024,256] bf16, ctx:[B,1024,256] bf16, Wout:[256,512] bf16 row-major [d][c]
// out: f32 [B,2048,256]; rows rowOff..rowOff+1023 of each batch belong to this stream.
__global__ void __launch_bounds__(128) attn_kernel(const bf16* __restrict__ Q,
                                                   const bf16* __restrict__ ctx,
                                                   const bf16* __restrict__ Wout,
                                                   float* __restrict__ out, int rowOff) {
  __shared__ bf16 kv_row[2][64 * 256] __attribute__((aligned(16))); // ctx chunk [s][d], dbl-buffered
  __shared__ bf16 kv_col[256 * 64] __attribute__((aligned(16)));    // ctx chunk [d][s]
  __shared__ bf16 p_lds[4][16 * 64] __attribute__((aligned(16)));   // per-wave P tile

  const int tid  = threadIdx.x;
  const int lane = tid & 31;
  const int wave = tid >> 5;
  const int bb   = blockIdx.y;
  const int qrow0 = blockIdx.x * 64 + wave * 16;   // [0,1024)

  const bf16* Qb = Q + ((size_t)bb * 1024 + qrow0) * 256;
  const bf16* cb = ctx + (size_t)bb * 1024 * 256;

  const unsigned kv0_off = (unsigned)(uintptr_t)&kv_row[0][0];
  const unsigned kv1_off = (unsigned)(uintptr_t)&kv_row[1][0];

  bf16x16 qfr[8];
#pragma unroll
  for (int kc = 0; kc < 8; ++kc) qfr[kc] = load_a_frag(Qb + kc * 32, 256, lane);

  v8f acc[16];
#pragma unroll
  for (int dt = 0; dt < 16; ++dt) acc[dt] = vzero();
  float mrun[8], lrun[8];
#pragma unroll
  for (int r = 0; r < 8; ++r) { mrun[r] = -3.0e38f; lrun[r] = 0.f; }

  const int moff = (lane & 16) ? 8 : 0;
  const int c16  = lane & 15;

  // kick off chunk 0 -> buffer 0 (each thread: 16 async b128, contiguous 128 elems)
  {
    const bf16* g = cb + (size_t)tid * 128;
    unsigned l = kv0_off + (unsigned)tid * 256;
#pragma unroll
    for (int j = 0; j < 16; ++j) async_ld_b128(l + j * 16, g + j * 8);
  }

  for (int ch = 0; ch < 16; ++ch) {   // 16 chunks of 64 context rows
    const int buf = ch & 1;
    __syncthreads();                  // prior compute done -> other buffer reusable
    if (ch + 1 < 16) {                // prefetch next chunk into other buffer
      const bf16* g = cb + (size_t)(ch + 1) * 64 * 256 + (size_t)tid * 128;
      unsigned l = (buf ? kv0_off : kv1_off) + (unsigned)tid * 256;
#pragma unroll
      for (int j = 0; j < 16; ++j) async_ld_b128(l + j * 16, g + j * 8);
      wait_async16();                 // current chunk's 16 async ops retired
    } else {
      wait_async0();
    }
    __syncthreads();                  // kv_row[buf] complete for all waves

    // build transposed copy kv_col[d][s] from kv_row[buf][s][d] (paired b32 stores)
    {
      const int rp = tid & 31;        // rows 2rp, 2rp+1
      const int cq = tid >> 5;        // 64-col quarter
#pragma unroll
      for (int j = 0; j < 8; ++j) {
        int c0 = cq * 64 + j * 8;
        bf16x8 v0 = *(const bf16x8*)&kv_row[buf][(2 * rp) * 256 + c0];
        bf16x8 v1 = *(const bf16x8*)&kv_row[buf][(2 * rp + 1) * 256 + c0];
#pragma unroll
        for (int e = 0; e < 8; ++e) {
          unsigned u = (unsigned)__builtin_bit_cast(unsigned short, v0[e])
                     | ((unsigned)__builtin_bit_cast(unsigned short, v1[e]) << 16);
          *(unsigned*)&kv_col[(c0 + e) * 64 + 2 * rp] = u;
        }
      }
    }
    __syncthreads();

    // scores: 16x64 tile = 4 N-tiles, K=256
    v8f s[4];
#pragma unroll
    for (int nt = 0; nt < 4; ++nt) {
      s[nt] = vzero();
#pragma unroll
      for (int kc = 0; kc < 8; ++kc)
        s[nt] = wmma_bf16(qfr[kc],
                          load_bt_frag(&kv_row[buf][nt * 16 * 256 + kc * 32], 256, lane), s[nt]);
    }

    // online softmax (rows in VGPR index, cols across 16-lane halves)
#pragma unroll
    for (int r = 0; r < 8; ++r) {
      float cm = fmaxf(fmaxf(s[0][r], s[1][r]), fmaxf(s[2][r], s[3][r]));
      cm = fmaxf(cm, __shfl_xor(cm, 1, 32));
      cm = fmaxf(cm, __shfl_xor(cm, 2, 32));
      cm = fmaxf(cm, __shfl_xor(cm, 4, 32));
      cm = fmaxf(cm, __shfl_xor(cm, 8, 32));
      float nm = fmaxf(mrun[r], cm);
      float sc = __expf(mrun[r] - nm);
      mrun[r] = nm;
      float rs = 0.f;
#pragma unroll
      for (int nt = 0; nt < 4; ++nt) {
        float p = __expf(s[nt][r] - nm);
        s[nt][r] = p;
        rs += p;
      }
      rs += __shfl_xor(rs, 1, 32);
      rs += __shfl_xor(rs, 2, 32);
      rs += __shfl_xor(rs, 4, 32);
      rs += __shfl_xor(rs, 8, 32);
      lrun[r] = lrun[r] * sc + rs;
#pragma unroll
      for (int dt = 0; dt < 16; ++dt) acc[dt][r] *= sc;
    }

    // P (D-layout) -> LDS -> A-layout, then mix += P @ V
#pragma unroll
    for (int nt = 0; nt < 4; ++nt)
#pragma unroll
      for (int r = 0; r < 8; ++r)
        p_lds[wave][(moff + r) * 64 + nt * 16 + c16] = f2bf(s[nt][r]);
    asm volatile("s_wait_dscnt 0" ::: "memory");
#pragma unroll
    for (int kc2 = 0; kc2 < 2; ++kc2) {
      bf16x16 pf = load_a_frag(&p_lds[wave][kc2 * 32], 64, lane);
#pragma unroll
      for (int dt = 0; dt < 16; ++dt)
        acc[dt] = wmma_bf16(pf, load_bt_frag(&kv_col[dt * 16 * 64 + kc2 * 32], 64, lane), acc[dt]);
    }
  }

  // normalize mix
#pragma unroll
  for (int r = 0; r < 8; ++r) {
    float inv = 1.0f / lrun[r];
#pragma unroll
    for (int dt = 0; dt < 16; ++dt) acc[dt][r] *= inv;
  }

  // epilogue: out = tanh(mix @ Wo1^T + q @ Wo2^T); reuse kv_row[0] per wave
  __syncthreads();
  bf16* mixb = &kv_row[0][wave * 16 * 256];
#pragma unroll
  for (int dt = 0; dt < 16; ++dt)
#pragma unroll
    for (int r = 0; r < 8; ++r)
      mixb[(moff + r) * 256 + dt * 16 + c16] = f2bf(acc[dt][r]);
  asm volatile("s_wait_dscnt 0" ::: "memory");

  float* ob = out + ((size_t)bb * 2048 + rowOff + qrow0) * 256;
  for (int nt = 0; nt < 16; ++nt) {
    v8f o = vzero();
#pragma unroll
    for (int kc = 0; kc < 8; ++kc)
      o = wmma_bf16(load_a_frag(mixb + kc * 32, 256, lane),
                    load_bt_frag(Wout + (size_t)(nt * 16) * 512 + kc * 32, 512, lane), o);
#pragma unroll
    for (int kc = 0; kc < 8; ++kc)
      o = wmma_bf16(qfr[kc],
                    load_bt_frag(Wout + (size_t)(nt * 16) * 512 + 256 + kc * 32, 512, lane), o);
#pragma unroll
    for (int r = 0; r < 8; ++r)
      ob[(size_t)(moff + r) * 256 + nt * 16 + c16] = tanhf(o[r]);
  }
}

// ---------------------------------------------------------------------- launch
extern "C" void kernel_launch(void* const* d_in, const int* in_sizes, int n_in,
                              void* d_out, int out_size, void* d_ws, size_t ws_size,
                              hipStream_t stream) {
  const size_t BLD = (size_t)64 * 1024 * 256;

  bf16* wsb   = (bf16*)d_ws;
  bf16* logb  = wsb;              // [B,L,D] bf16
  bf16* metb  = logb + BLD;
  bf16* Qa    = metb + BLD;
  bf16* Qb    = Qa + BLD;
  bf16* WinA  = Qb + BLD;         // [256,256]
  bf16* WinB  = WinA + 256 * 256;
  bf16* WoutA = WinB + 256 * 256; // [256,512]
  bf16* WoutB = WoutA + 256 * 512;

  auto cast = [&](int idx, bf16* dst, size_t n) {
    int n4 = (int)(n / 4);
    cast_bf16_kernel<<<(n4 + 255) / 256, 256, 0, stream>>>((const float*)d_in[idx], dst, n4);
  };
  cast(0, logb, BLD);
  cast(1, metb, BLD);
  cast(2, WinA, 256 * 256);
  cast(4, WinB, 256 * 256);
  cast(3, WoutA, 256 * 512);
  cast(5, WoutB, 256 * 512);

  // Q projections: M = 64*1024 rows, 32 rows per wave, 4 waves per block
  qproj_kernel<<<512, 128, 0, stream>>>(logb, WinA, Qa);
  qproj_kernel<<<512, 128, 0, stream>>>(metb, WinB, Qb);

  // fused attention + output projection, one launch per stream
  attn_kernel<<<dim3(16, 64), 128, 0, stream>>>(Qa, metb, WoutA, (float*)d_out, 0);
  attn_kernel<<<dim3(16, 64), 128, 0, stream>>>(Qb, logb, WoutB, (float*)d_out, 1024);
}
